// Classifier_temporal_5729486373631
// MI455X (gfx1250) — matrix operand order repaired and checked by reading, not compiled, not run on hardware
//
#include <hip/hip_runtime.h>
#include <hip/hip_bf16.h>

typedef __bf16 bf16_t;
typedef bf16_t bf16x4  __attribute__((ext_vector_type(4)));
typedef bf16_t bf16x8  __attribute__((ext_vector_type(8)));
typedef bf16_t bf16x16 __attribute__((ext_vector_type(16)));
typedef float  f32x8   __attribute__((ext_vector_type(8)));

#define C_PAD  160   // 129 input channels padded to 5 * 32 (WMMA K tiles)
#define G_COLS 192   // 3 gates * 64
#define OUTC   64

__device__ __forceinline__ float sigmoid_f(float x) { return 1.0f / (1.0f + __expf(-x)); }

// native no-return f32 atomic add (global_atomic_add_f32), relaxed, device scope
__device__ __forceinline__ void atomic_fadd(float* p, float v) {
  (void)__hip_atomic_fetch_add(p, v, __ATOMIC_RELAXED, __HIP_MEMORY_SCOPE_AGENT);
}

// ----------------------------------------------------------------------------
// Setup kernels
// ----------------------------------------------------------------------------
__global__ void k_init_deg(float* deg, int N) {
  int i = blockIdx.x * blockDim.x + threadIdx.x;
  if (i < N) deg[i] = 1.0f;  // self loop
}

__global__ void k_deg(const int* __restrict__ ei, float* __restrict__ deg, int E) {
  int e = blockIdx.x * blockDim.x + threadIdx.x;
  if (e < E) atomic_fadd(&deg[ei[E + e]], 1.0f);
}

__global__ void k_dinv(const float* __restrict__ deg, float* __restrict__ dinv,
                       float* __restrict__ selfn, int N) {
  int i = blockIdx.x * blockDim.x + threadIdx.x;
  if (i >= N) return;
  float d = __frsqrt_rn(deg[i]);
  dinv[i] = d;
  selfn[i] = d * d;
}

__global__ void k_en(const int* __restrict__ ei, const float* __restrict__ dinv,
                     float* __restrict__ en, int E) {
  int e = blockIdx.x * blockDim.x + threadIdx.x;
  if (e < E) en[e] = dinv[ei[e]] * dinv[ei[E + e]];
}

// x (N,129,T) fp32 -> xt (T,N,160) bf16, zero-padded channels
__global__ void k_transpose(const float* __restrict__ x, bf16_t* __restrict__ xt,
                            int N, int T) {
  int i = blockIdx.x * blockDim.x + threadIdx.x;
  if (i >= N * C_PAD) return;
  int c = i % C_PAD;
  int n = i / C_PAD;
  bf16_t* dst = xt + i;
  size_t tstride = (size_t)N * C_PAD;
  if (c < 129) {
    const float* s = x + ((size_t)n * 129 + c) * T;
    for (int t = 0; t < T; ++t) dst[(size_t)t * tstride] = (bf16_t)s[t];
  } else {
    for (int t = 0; t < T; ++t) dst[(size_t)t * tstride] = (bf16_t)0.0f;
  }
}

// Pack a B[k][n] element into the WMMA bf16 B-fragment layout:
// frag (kt,nt); lane = (n%16) + 16*((k%32)>=16); element j = k%16.
__device__ __forceinline__ void packB(bf16_t* Bp, int NT, int k, int n, float v) {
  int lane = (n & 15) + (((k & 31) >= 16) ? 16 : 0);
  Bp[(size_t)((((k >> 5) * NT) + (n >> 4)) * 32 + lane) * 16 + (k & 15)] = (bf16_t)v;
}

__global__ void k_pack_consts(
    const float* __restrict__ Wz, const float* __restrict__ Wr, const float* __restrict__ Wh,
    const float* __restrict__ bz, const float* __restrict__ br, const float* __restrict__ bh,
    const float* __restrict__ Wlz, const float* __restrict__ Wlr, const float* __restrict__ Wlh,
    const float* __restrict__ blz, const float* __restrict__ blr, const float* __restrict__ blh,
    const float* __restrict__ att, int T,
    bf16_t* __restrict__ B1p, bf16_t* __restrict__ B2p, bf16_t* __restrict__ B3p,
    float* __restrict__ bias192, float* __restrict__ bl2, float* __restrict__ blh_o,
    float* __restrict__ probs) {
  int tid = threadIdx.x;
  // B1: (C_PAD x 192) = [W_z | W_r | W_h], zero-padded K
  for (int i = tid; i < C_PAD * 192; i += blockDim.x) {
    int k = i / 192, n = i % 192;
    int g = n >> 6, nn = n & 63;
    const float* W = (g == 0) ? Wz : (g == 1) ? Wr : Wh;
    float v = (k < 129) ? W[k * 64 + nn] : 0.0f;
    packB(B1p, 12, k, n, v);
  }
  // B2: (192 x 128), block structured so [G_z|G_r|H] @ B2 = [Z_pre|R_pre]
  for (int i = tid; i < 192 * 128; i += blockDim.x) {
    int k = i / 128, n = i % 128;
    float v = 0.0f;
    if (n < 64) {
      if (k < 64) v = Wlz[k * 64 + n];
      else if (k >= 128) v = Wlz[(k - 64) * 64 + n];
    } else {
      int nn = n - 64;
      if (k >= 64) v = Wlr[(k - 64) * 64 + nn];
    }
    packB(B2p, 8, k, n, v);
  }
  // B3: (128 x 64) = Wl_h
  for (int i = tid; i < 128 * 64; i += blockDim.x) {
    int k = i / 64, n = i % 64;
    packB(B3p, 4, k, n, Wlh[k * 64 + n]);
  }
  for (int i = tid; i < 192; i += blockDim.x)
    bias192[i] = (i < 64) ? bz[i] : (i < 128) ? br[i - 64] : bh[i - 128];
  for (int i = tid; i < 128; i += blockDim.x)
    bl2[i] = (i < 64) ? blz[i] : blr[i - 64];
  for (int i = tid; i < 64; i += blockDim.x) blh_o[i] = blh[i];
  if (tid == 0) {  // softmax(attention)
    float mx = att[0];
    for (int t = 1; t < T; ++t) mx = fmaxf(mx, att[t]);
    float s = 0.0f;
    for (int t = 0; t < T; ++t) s += __expf(att[t] - mx);
    for (int t = 0; t < T; ++t) probs[t] = __expf(att[t] - mx) / s;
  }
}

// ----------------------------------------------------------------------------
// Per-step elementwise / scatter kernels
// ----------------------------------------------------------------------------
__global__ void k_init_xa(const bf16_t* __restrict__ xt_t, const float* __restrict__ selfn,
                          float* __restrict__ xa, int total) {
  int i = blockIdx.x * blockDim.x + threadIdx.x;
  if (i >= total) return;
  int n = i / C_PAD;
  xa[i] = selfn[n] * (float)xt_t[i];
}

// one wave per edge: lanes cover channels [lane*4, lane*4+4); lane0 adds ch 128
__global__ __launch_bounds__(256) void k_scatter(
    const bf16_t* __restrict__ xt_t, const int* __restrict__ ei,
    const float* __restrict__ en, float* __restrict__ xa, int E) {
  size_t gtid = (size_t)blockIdx.x * blockDim.x + threadIdx.x;
  int e = (int)(gtid >> 5);
  if (e >= E) return;
  int lane = threadIdx.x & 31;
  int src = ei[e];
  int dst = ei[E + e];
  float w = en[e];
  const bf16_t* xr = xt_t + (size_t)src * C_PAD;
  float* ar = xa + (size_t)dst * C_PAD;
  int c = lane * 4;
  bf16x4 v = *(const bf16x4*)(xr + c);
  atomic_fadd(&ar[c + 0], w * (float)v[0]);
  atomic_fadd(&ar[c + 1], w * (float)v[1]);
  atomic_fadd(&ar[c + 2], w * (float)v[2]);
  atomic_fadd(&ar[c + 3], w * (float)v[3]);
  if (lane == 0) {  // channel 128 (129..159 are zero pad)
    atomic_fadd(&ar[128], w * (float)xr[128]);
  }
}

__global__ void k_cvt_xa(const float* __restrict__ xa, bf16_t* __restrict__ A1, int total) {
  int i = blockIdx.x * blockDim.x + threadIdx.x;
  if (i < total) A1[i] = (bf16_t)xa[i];
}

// A2[:,128:192] = bf16(H)   (A2[:,0:128] is written by GEMM1's epilogue)
__global__ void k_pack_H(const float* __restrict__ H, bf16_t* __restrict__ A2, int total) {
  int i = blockIdx.x * blockDim.x + threadIdx.x;  // total = N*64
  if (i >= total) return;
  int c = i & 63;
  int n = i >> 6;
  A2[(size_t)n * G_COLS + 128 + c] = (bf16_t)H[i];
}

// A3[:,64:128] = bf16(H * R)   (A3[:,0:64] is written by GEMM1's epilogue)
__global__ void k_pack_HR(const float* __restrict__ H, const float* __restrict__ ZR,
                          bf16_t* __restrict__ A3, int total) {
  int i = blockIdx.x * blockDim.x + threadIdx.x;  // total = N*64
  if (i >= total) return;
  int c = i & 63;
  int n = i >> 6;
  A3[(size_t)n * 128 + 64 + c] = (bf16_t)(H[i] * ZR[(size_t)n * 128 + 64 + c]);
}

// ----------------------------------------------------------------------------
// WMMA GEMM core: each wave owns a 16-row M tile x (NT*16) columns.
// A: row-major bf16 (stride KT*32). B: prepacked fragments staged in LDS.
// ----------------------------------------------------------------------------
__device__ __forceinline__ void fill_lds_frags(bf16x16* lds, const bf16_t* g, int nfrag) {
  uint4* d = (uint4*)lds;
  const uint4* s = (const uint4*)g;
  int n16 = nfrag * 2;  // 32B per fragment slot
  for (int i = threadIdx.x; i < n16; i += blockDim.x) d[i] = s[i];
  __syncthreads();
}

template <int KT, int NT>
__device__ __forceinline__ void wmma_rowtile(const bf16_t* __restrict__ A,
                                             const bf16x16* __restrict__ Blds,
                                             int row0, f32x8* acc) {
  const int lane = threadIdx.x & 31;
  const int m = row0 + (lane & 15);
  const int k0 = (lane >> 4) * 16;  // A frag: lane<16 holds K {0..15}, lane>=16 {16..31}
  const size_t K = (size_t)KT * 32;
#pragma unroll
  for (int nt = 0; nt < NT; ++nt)
#pragma unroll
    for (int j = 0; j < 8; ++j) acc[nt][j] = 0.0f;
#pragma unroll
  for (int kt = 0; kt < KT; ++kt) {
    const bf16_t* ap = A + (size_t)m * K + kt * 32 + k0;
    bf16x8 alo = *(const bf16x8*)(ap);
    bf16x8 ahi = *(const bf16x8*)(ap + 8);
    bf16x16 af = __builtin_shufflevector(alo, ahi, 0, 1, 2, 3, 4, 5, 6, 7, 8, 9, 10, 11, 12, 13, 14, 15);
#pragma unroll
    for (int nt = 0; nt < NT; ++nt) {
      bf16x16 bfr = Blds[(kt * NT + nt) * 32 + lane];
      acc[nt] = __builtin_amdgcn_wmma_f32_16x16x32_bf16(
          false, af, false, bfr, (short)0, acc[nt], false, false);
    }
  }
}

// GEMM1: A1 (N x 160) @ B1 (160 x 192) + bias, emitted as bf16 straight into
// the downstream A matrices: cols 0..127 -> A2 (G_z|G_r), cols 128..191 -> A3 (G_h)
__global__ __launch_bounds__(256) void k_gemm_xw(
    const bf16_t* __restrict__ A, const bf16_t* __restrict__ Bp,
    const float* __restrict__ bias,
    bf16_t* __restrict__ A2, bf16_t* __restrict__ A3, int nMt) {
  __shared__ bf16x16 lb[5 * 12 * 32];
  fill_lds_frags(lb, Bp, 5 * 12 * 32);
  int mtile = blockIdx.x * 8 + (threadIdx.x >> 5);
  if (mtile >= nMt) return;
  f32x8 acc[12];
  wmma_rowtile<5, 12>(A, lb, mtile * 16, acc);
  int lane = threadIdx.x & 31;
  int nl = lane & 15, mo = (lane >> 4) << 3;
#pragma unroll
  for (int nt = 0; nt < 12; ++nt) {
    int n = nt * 16 + nl;
    float b = bias[n];
#pragma unroll
    for (int v = 0; v < 8; ++v) {
      int m = mtile * 16 + mo + v;
      float val = acc[nt][v] + b;
      if (nt < 8) A2[(size_t)m * G_COLS + n] = (bf16_t)val;          // G_z | G_r
      else        A3[(size_t)m * 128 + (n - 128)] = (bf16_t)val;     // G_h
    }
  }
}

// GEMM2: [G_z|G_r|H] (N x 192) @ B2 (192 x 128) -> sigmoid -> ZR
__global__ __launch_bounds__(256) void k_gemm_zr(
    const bf16_t* __restrict__ A, const bf16_t* __restrict__ Bp,
    const float* __restrict__ bl2, float* __restrict__ ZR, int nMt) {
  __shared__ bf16x16 lb[6 * 8 * 32];
  fill_lds_frags(lb, Bp, 6 * 8 * 32);
  int mtile = blockIdx.x * 8 + (threadIdx.x >> 5);
  if (mtile >= nMt) return;
  f32x8 acc[8];
  wmma_rowtile<6, 8>(A, lb, mtile * 16, acc);
  int lane = threadIdx.x & 31;
  int nl = lane & 15, mo = (lane >> 4) << 3;
#pragma unroll
  for (int nt = 0; nt < 8; ++nt) {
    int n = nt * 16 + nl;
    float b = bl2[n];
#pragma unroll
    for (int v = 0; v < 8; ++v) {
      int m = mtile * 16 + mo + v;
      ZR[(size_t)m * 128 + n] = sigmoid_f(acc[nt][v] + b);
    }
  }
}

// GEMM3: [G_h|H*R] (N x 128) @ Wl_h (128 x 64) -> fused tanh + GRU + attention accum
__global__ __launch_bounds__(256) void k_gemm_h(
    const bf16_t* __restrict__ A, const bf16_t* __restrict__ Bp,
    const float* __restrict__ blh, const float* __restrict__ ZR,
    const float* __restrict__ probs, int t,
    float* __restrict__ H, float* __restrict__ Hacc, int nMt) {
  __shared__ bf16x16 lb[4 * 4 * 32];
  fill_lds_frags(lb, Bp, 4 * 4 * 32);
  int mtile = blockIdx.x * 8 + (threadIdx.x >> 5);
  if (mtile >= nMt) return;
  f32x8 acc[4];
  wmma_rowtile<4, 4>(A, lb, mtile * 16, acc);
  float p = probs[t];
  int lane = threadIdx.x & 31;
  int nl = lane & 15, mo = (lane >> 4) << 3;
#pragma unroll
  for (int nt = 0; nt < 4; ++nt) {
    int n = nt * 16 + nl;
    float b = blh[n];
#pragma unroll
    for (int v = 0; v < 8; ++v) {
      int m = mtile * 16 + mo + v;
      float ht = tanhf(acc[nt][v] + b);
      float z = ZR[(size_t)m * 128 + n];
      size_t hi = (size_t)m * OUTC + n;
      float hn = z * H[hi] + (1.0f - z) * ht;
      H[hi] = hn;
      Hacc[hi] += p * hn;
    }
  }
}

// ----------------------------------------------------------------------------
// Output: y_pred = sigmoid(Hacc @ W_o + b_o)[train_idx], then y[train_idx]
// ----------------------------------------------------------------------------
__global__ void k_out(const float* __restrict__ Hacc, const float* __restrict__ Wo,
                      const float* __restrict__ bo, const int* __restrict__ tidx,
                      const float* __restrict__ y, float* __restrict__ out, int ntrain) {
  int i = blockIdx.x * blockDim.x + threadIdx.x;
  if (i >= ntrain) return;
  int n = tidx[i];
  float s = bo[0];
  const float* h = Hacc + (size_t)n * OUTC;
#pragma unroll 8
  for (int c = 0; c < OUTC; ++c) s += h[c] * Wo[c];
  out[i] = sigmoid_f(s);
  out[ntrain + i] = y[n];
}

// ----------------------------------------------------------------------------
extern "C" void kernel_launch(void* const* d_in, const int* in_sizes, int n_in,
                              void* d_out, int out_size, void* d_ws, size_t ws_size,
                              hipStream_t stream) {
  (void)n_in; (void)out_size; (void)ws_size;
  const float* x    = (const float*)d_in[0];
  const int*   ei   = (const int*)d_in[1];
  const float* y    = (const float*)d_in[2];
  const int*   tidx = (const int*)d_in[3];
  const float* Wz   = (const float*)d_in[4];
  const float* bz   = (const float*)d_in[5];
  const float* Wlz  = (const float*)d_in[6];
  const float* blz  = (const float*)d_in[7];
  const float* Wr   = (const float*)d_in[8];
  const float* br   = (const float*)d_in[9];
  const float* Wlr  = (const float*)d_in[10];
  const float* blr  = (const float*)d_in[11];
  const float* Wh   = (const float*)d_in[12];
  const float* bh   = (const float*)d_in[13];
  const float* Wlh  = (const float*)d_in[14];
  const float* blh  = (const float*)d_in[15];
  const float* att  = (const float*)d_in[16];
  const float* Wo   = (const float*)d_in[17];
  const float* bo   = (const float*)d_in[18];

  const int N      = in_sizes[2];        // 50000
  const int E      = in_sizes[1] / 2;    // 1600000
  const int T      = in_sizes[16];       // 25
  const int NTRAIN = in_sizes[3];        // 25000
  const int nMt    = N / 16;             // 3125 M-tiles

  // workspace arena (256B aligned slots)
  char* ws = (char*)d_ws;
  size_t off = 0;
  auto alloc = [&](size_t bytes) {
    char* p = ws + off;
    off = (off + bytes + 255) & ~(size_t)255;
    return p;
  };
  bf16_t* xt      = (bf16_t*)alloc((size_t)T * N * C_PAD * sizeof(bf16_t));
  float*  xa      = (float*)alloc((size_t)N * C_PAD * sizeof(float));
  bf16_t* A1      = (bf16_t*)alloc((size_t)N * C_PAD * sizeof(bf16_t));
  bf16_t* A2      = (bf16_t*)alloc((size_t)N * G_COLS * sizeof(bf16_t));
  float*  ZR      = (float*)alloc((size_t)N * 128 * sizeof(float));
  bf16_t* A3      = (bf16_t*)alloc((size_t)N * 128 * sizeof(bf16_t));
  float*  H       = (float*)alloc((size_t)N * OUTC * sizeof(float));
  float*  Hacc    = (float*)alloc((size_t)N * OUTC * sizeof(float));
  float*  deg     = (float*)alloc((size_t)N * sizeof(float));
  float*  dinv    = (float*)alloc((size_t)N * sizeof(float));
  float*  selfn   = (float*)alloc((size_t)N * sizeof(float));
  float*  en      = (float*)alloc((size_t)E * sizeof(float));
  bf16_t* B1p     = (bf16_t*)alloc((size_t)5 * 12 * 32 * 16 * sizeof(bf16_t));
  bf16_t* B2p     = (bf16_t*)alloc((size_t)6 * 8 * 32 * 16 * sizeof(bf16_t));
  bf16_t* B3p     = (bf16_t*)alloc((size_t)4 * 4 * 32 * 16 * sizeof(bf16_t));
  float*  bias192 = (float*)alloc(192 * sizeof(float));
  float*  bl2     = (float*)alloc(128 * sizeof(float));
  float*  blh2    = (float*)alloc(64 * sizeof(float));
  float*  probs   = (float*)alloc(32 * sizeof(float));

  (void)hipMemsetAsync(H, 0, (size_t)N * OUTC * sizeof(float), stream);
  (void)hipMemsetAsync(Hacc, 0, (size_t)N * OUTC * sizeof(float), stream);

  k_init_deg<<<(N + 255) / 256, 256, 0, stream>>>(deg, N);
  k_deg<<<(E + 255) / 256, 256, 0, stream>>>(ei, deg, E);
  k_dinv<<<(N + 255) / 256, 256, 0, stream>>>(deg, dinv, selfn, N);
  k_en<<<(E + 255) / 256, 256, 0, stream>>>(ei, dinv, en, E);
  k_transpose<<<(N * C_PAD + 255) / 256, 256, 0, stream>>>(x, xt, N, T);
  k_pack_consts<<<1, 256, 0, stream>>>(Wz, Wr, Wh, bz, br, bh, Wlz, Wlr, Wlh,
                                       blz, blr, blh, att, T,
                                       B1p, B2p, B3p, bias192, bl2, blh2, probs);

  const int tot1 = N * C_PAD, totH = N * OUTC;
  const int gemmBlocks = (nMt + 7) / 8;
  const int scatBlocks = (int)(((size_t)E * 32 + 255) / 256);

  for (int t = 0; t < T; ++t) {
    const bf16_t* xt_t = xt + (size_t)t * N * C_PAD;
    k_init_xa<<<(tot1 + 255) / 256, 256, 0, stream>>>(xt_t, selfn, xa, tot1);
    k_scatter<<<scatBlocks, 256, 0, stream>>>(xt_t, ei, en, xa, E);
    k_cvt_xa<<<(tot1 + 255) / 256, 256, 0, stream>>>(xa, A1, tot1);
    k_gemm_xw<<<gemmBlocks, 256, 0, stream>>>(A1, B1p, bias192, A2, A3, nMt);
    k_pack_H<<<(totH + 255) / 256, 256, 0, stream>>>(H, A2, totH);
    k_gemm_zr<<<gemmBlocks, 256, 0, stream>>>(A2, B2p, bl2, ZR, nMt);
    k_pack_HR<<<(totH + 255) / 256, 256, 0, stream>>>(H, ZR, A3, totH);
    k_gemm_h<<<gemmBlocks, 256, 0, stream>>>(A3, B3p, blh2, ZR, probs, t, H, Hacc, nMt);
  }

  k_out<<<(NTRAIN + 255) / 256, 256, 0, stream>>>(Hacc, Wo, bo, tidx, y,
                                                  (float*)d_out, NTRAIN);
}
